// BalancedConformationalConsistencyLoss_15607911153863
// MI455X (gfx1250) — matrix operand order repaired and checked by reading, not compiled, not run on hardware
//
#include <hip/hip_runtime.h>

typedef __bf16 bf16_t;
typedef __attribute__((ext_vector_type(16))) __bf16 v16bf;
typedef __attribute__((ext_vector_type(8)))  float  v8f;

#define NN    8192
#define DD    256
#define NFRG  256
#define TDIM  512
#define NVEC  (NN*3)
#define ROWCAP 96

__device__ __forceinline__ float ssp_f(float x) {
  float sp = (x > 15.0f) ? x : log1pf(__expf(x));
  return sp - 0.69314718056f;
}

__device__ __forceinline__ float block_reduce256(float v, float* red) {
  const int t = threadIdx.x;
  red[t] = v;
  __syncthreads();
  #pragma unroll
  for (int s = 128; s > 0; s >>= 1) {
    if (t < s) red[t] += red[t + s];
    __syncthreads();
  }
  float r = red[0];
  __syncthreads();
  return r;
}

// CDNA5 async global->LDS copy (16B), tracked by ASYNCcnt.
// LDS address = low 32 bits of the generic pointer.
__device__ __forceinline__ void async_b128(void* lds, const void* g) {
  unsigned l = (unsigned)(unsigned long long)lds;
  asm volatile("global_load_async_to_lds_b128 %0, %1, off"
               :
               : "v"(l), "v"(g)
               : "memory");
}

__device__ __forceinline__ void wait_async0() {
  asm volatile("s_wait_asynccnt 0x0" ::: "memory");
}

// ---------------------------------------------------------------------------
// Elementwise prep
// ---------------------------------------------------------------------------
__global__ __launch_bounds__(256) void k_lerp_bf16(const float* __restrict__ a,
                                                   const float* __restrict__ b,
                                                   bf16_t* __restrict__ o,
                                                   int n, float r) {
  int i = blockIdx.x * blockDim.x + threadIdx.x;
  if (i < n) o[i] = (bf16_t)(a[i] * r + b[i] * (1.0f - r));
}

// W: K x N fp32 row-major  ->  Wt: N x K bf16 row-major (transposed)
__global__ __launch_bounds__(256) void k_cvt_transpose_bf16(
    const float* __restrict__ W, bf16_t* __restrict__ Wt, int K, int N) {
  int i = blockIdx.x * blockDim.x + threadIdx.x;
  if (i < K * N) {
    int k = i / N, n = i - k * N;
    Wt[(size_t)n * K + k] = (bf16_t)W[i];
  }
}

// ---------------------------------------------------------------------------
// Tiled WMMA GEMM:  C = [ssp](A @ B + bias)
// A : M x K bf16 row-major (activations)
// Bt: N x K bf16 row-major (pre-transposed weights)
// Block = 256 threads = 8 waves; 64x128 C tile; wave computes 16x64 (4 WMMAs).
// Double-buffered LDS staging via global_load_async_to_lds_b128 (ASYNCcnt).
// M mult of 64, N mult of 128, K mult of 32.
// ---------------------------------------------------------------------------
__global__ __launch_bounds__(256) void gemm_bf16_wmma(
    const bf16_t* __restrict__ A, const bf16_t* __restrict__ Bt,
    const float* __restrict__ bias, bf16_t* __restrict__ Cb,
    float* __restrict__ Cf, int M, int Nn, int K, int do_ssp) {
  __shared__ bf16_t As[2][64][40];    // [m][k], stride 40 (80B, 16B-aligned)
  __shared__ bf16_t Bs[2][128][40];   // [n][k]

  const int tid  = threadIdx.x;
  const int lane = tid & 31;
  const int wave = tid >> 5;
  const int wm   = wave & 3;          // row tile (16 rows)
  const int wn   = wave >> 2;         // col half (64 cols)
  const int lrow = lane & 15;
  const int hi   = lane >> 4;

  const int bn0 = blockIdx.x * 128;
  const int bm0 = blockIdx.y * 64;

  const int arow = tid >> 2;          // 0..63
  const int acol = (tid & 3) * 8;     // 0,8,16,24
  const int brow = tid >> 1;          // 0..127
  const int bcol = (tid & 1) * 16;    // 0,16

  v8f acc[4] = {};

  auto issue_tile = [&](int k0, int buf) {
    async_b128(&As[buf][arow][acol],
               &A[(size_t)(bm0 + arow) * K + k0 + acol]);
    async_b128(&Bs[buf][brow][bcol],
               &Bt[(size_t)(bn0 + brow) * K + k0 + bcol]);
    async_b128(&Bs[buf][brow][bcol + 8],
               &Bt[(size_t)(bn0 + brow) * K + k0 + bcol + 8]);
  };

  int buf = 0;
  issue_tile(0, 0);
  wait_async0();
  __syncthreads();

  for (int k0 = 0; k0 < K; k0 += 32) {
    const bool more = (k0 + 32) < K;
    if (more) issue_tile(k0 + 32, buf ^ 1);

    // A fragment: lane half 'hi' holds K = {hi*8..hi*8+7, hi*8+16..hi*8+23}
    v16bf af;
    #pragma unroll
    for (int i = 0; i < 8; ++i) af[i] = As[buf][wm * 16 + lrow][hi * 8 + i];
    #pragma unroll
    for (int i = 0; i < 8; ++i) af[8 + i] = As[buf][wm * 16 + lrow][hi * 8 + 16 + i];

    // B fragments: lane half 'hi' holds K = {hi*8..hi*8+7, hi*8+16..hi*8+23} for col lrow
    v16bf bft[4];
    #pragma unroll
    for (int t = 0; t < 4; ++t) {
      const bf16_t* bp = &Bs[buf][wn * 64 + t * 16 + lrow][hi * 8];
      #pragma unroll
      for (int i = 0; i < 16; ++i) bft[t][i] = bp[(i < 8) ? i : (i + 8)];
    }

    #pragma unroll
    for (int t = 0; t < 4; ++t)
      acc[t] = __builtin_amdgcn_wmma_f32_16x16x32_bf16(
          false, af, false, bft[t], (short)0, acc[t], false, false);

    if (more) wait_async0();
    __syncthreads();
    buf ^= 1;
  }

  // Epilogue: C element (r, lane): M = r + 8*hi, N = lrow within a 16x16 tile
  #pragma unroll
  for (int t = 0; t < 4; ++t) {
    const int col = bn0 + wn * 64 + t * 16 + lrow;
    const float bv = bias ? bias[col] : 0.0f;
    #pragma unroll
    for (int r = 0; r < 8; ++r) {
      const int row = bm0 + wm * 16 + hi * 8 + r;
      float v = acc[t][r] + bv;
      if (do_ssp) v = ssp_f(v);
      if (Cb) Cb[(size_t)row * Nn + col] = (bf16_t)v;
      if (Cf) Cf[(size_t)row * Nn + col] = v;
    }
  }
}

// ---------------------------------------------------------------------------
// Row L2-normalize (row length 256), one wave per row.
// Optionally accumulates sum of squares (Frobenius) for l2_reg.
// ---------------------------------------------------------------------------
__global__ __launch_bounds__(256) void normalize_rows256(
    const float* __restrict__ src, long src_stride,
    bf16_t* __restrict__ dst, long dst_stride, int rows,
    float* __restrict__ accum) {
  const int lane = threadIdx.x & 31;
  const int wave = threadIdx.x >> 5;
  const int row  = blockIdx.x * 8 + wave;
  if (row >= rows) return;
  const float* s = src + (size_t)row * src_stride;
  float x[8];
  float ss = 0.0f;
  #pragma unroll
  for (int j = 0; j < 8; ++j) { x[j] = s[lane + 32 * j]; ss += x[j] * x[j]; }
  #pragma unroll
  for (int m = 16; m >= 1; m >>= 1) ss += __shfl_xor(ss, m, 32);
  if (accum && lane == 0) atomicAdd(accum, ss);
  const float inv = 1.0f / fmaxf(sqrtf(ss), 1e-12f);
  bf16_t* d = dst + (size_t)row * dst_stride;
  #pragma unroll
  for (int j = 0; j < 8; ++j) d[lane + 32 * j] = (bf16_t)(x[j] * inv);
}

// ---------------------------------------------------------------------------
// Fragment bucketing
// ---------------------------------------------------------------------------
__global__ __launch_bounds__(256) void k_count(const int* __restrict__ frag,
                                               int* __restrict__ counts) {
  int i = blockIdx.x * blockDim.x + threadIdx.x;
  if (i < NN) atomicAdd(&counts[frag[i]], 1);
}

__global__ __launch_bounds__(256) void k_scan(const int* __restrict__ counts,
                                              int* __restrict__ offs,
                                              int* __restrict__ cursor) {
  __shared__ int tmp[NFRG];
  const int t = threadIdx.x;
  tmp[t] = counts[t];
  __syncthreads();
  for (int d = 1; d < NFRG; d <<= 1) {
    int v = (t >= d) ? tmp[t - d] : 0;
    __syncthreads();
    tmp[t] += v;
    __syncthreads();
  }
  offs[t]   = tmp[t] - counts[t];
  cursor[t] = tmp[t] - counts[t];
  if (t == NFRG - 1) offs[NFRG] = tmp[t];
}

__global__ __launch_bounds__(256) void k_scatter(const int* __restrict__ frag,
                                                 int* __restrict__ cursor,
                                                 int* __restrict__ members) {
  int i = blockIdx.x * blockDim.x + threadIdx.x;
  if (i < NN) {
    int p = atomicAdd(&cursor[frag[i]], 1);
    members[p] = i;
  }
}

// ---------------------------------------------------------------------------
// Per-fragment variance: var = (sum||x||^2 - ||sum x||^2/c) / c
// ---------------------------------------------------------------------------
__global__ __launch_bounds__(256) void frag_stats(
    const float* __restrict__ dec, const float* __restrict__ vinv,
    const int* __restrict__ members, const int* __restrict__ offs,
    const int* __restrict__ counts, float* __restrict__ sh_var,
    float* __restrict__ v_var) {
  __shared__ float red[256];
  const int f = blockIdx.x, t = threadIdx.x;
  const int cnt = counts[f], off = offs[f];
  float ssum = 0.f, v0 = 0.f, v1 = 0.f, v2 = 0.f, sqs = 0.f, sqv = 0.f;
  for (int m = 0; m < cnt; ++m) {
    const int idx = members[off + m];
    float x = dec[(size_t)idx * TDIM + t];
    ssum += x; sqs += x * x;
    float a = vinv[(size_t)idx * 768 + t];
    float b = vinv[(size_t)idx * 768 + 256 + t];
    float c = vinv[(size_t)idx * 768 + 512 + t];
    v0 += a; v1 += b; v2 += c;
    sqv += a * a + b * b + c * c;
  }
  float sumsq_s = block_reduce256(sqs, red);
  float s2      = block_reduce256(ssum * ssum, red);
  float sumsq_v = block_reduce256(sqv, red);
  float sv2     = block_reduce256(v0 * v0 + v1 * v1 + v2 * v2, red);
  if (t == 0) {
    float sc = fmaxf((float)cnt, 1.0f);
    sh_var[f] = (sumsq_s - s2 / sc) / sc;
    v_var[f]  = (sumsq_v - sv2 / sc) / sc;
  }
}

// ---------------------------------------------------------------------------
// Per-fragment pairwise MSE over 5 normalized matrices (LDS-staged rows).
// ---------------------------------------------------------------------------
__global__ __launch_bounds__(256) void frag_pairs(
    const bf16_t* __restrict__ shn, const bf16_t* __restrict__ unn,
    const bf16_t* __restrict__ vinn, const int* __restrict__ members,
    const int* __restrict__ offs, const int* __restrict__ counts,
    const int* __restrict__ atypes, float* __restrict__ sh_sim,
    float* __restrict__ un_sim, float* __restrict__ dir_loss) {
  __shared__ bf16_t rows[ROWCAP * DD];
  __shared__ float red[256];
  __shared__ float res[5];
  const int f = blockIdx.x, t = threadIdx.x;
  const int cnt = counts[f], off = offs[f];
  const int P = (cnt * (cnt - 1)) >> 1;
  auto Tfun = [cnt] __device__ (int a) { return (a * (2 * cnt - a - 1)) >> 1; };

  for (int mat = 0; mat < 5; ++mat) {
    const int nc = (cnt < ROWCAP) ? cnt : ROWCAP;
    for (int e = t; e < nc * DD; e += 256) {
      const int r = e >> 8, c = e & 255;
      const int g = members[off + r];
      const bf16_t* src = (mat == 0) ? (shn + (size_t)g * DD)
                        : (mat == 1) ? (unn + (size_t)g * DD)
                                     : (vinn + ((size_t)g * 3 + (mat - 2)) * DD);
      rows[e] = src[c];
    }
    __syncthreads();

    float lsum = 0.0f;
    for (int p = t; p < P; p += 256) {
      int i;
      {
        float fm = (float)cnt;
        float tt = 2.0f * fm - 1.0f;
        int ii = (int)floorf((tt - sqrtf(fmaxf(tt * tt - 8.0f * (float)p, 0.f))) * 0.5f);
        if (ii < 0) ii = 0;
        if (ii > cnt - 2) ii = cnt - 2;
        while (ii > 0 && Tfun(ii) > p) --ii;
        while (Tfun(ii + 1) <= p) ++ii;
        i = ii;
      }
      const int j = i + 1 + (p - Tfun(i));

      const int gi = members[off + i];
      const int gj = members[off + j];
      const bf16_t* gsi = (mat == 0) ? (shn + (size_t)gi * DD)
                        : (mat == 1) ? (unn + (size_t)gi * DD)
                                     : (vinn + ((size_t)gi * 3 + (mat - 2)) * DD);
      const bf16_t* gsj = (mat == 0) ? (shn + (size_t)gj * DD)
                        : (mat == 1) ? (unn + (size_t)gj * DD)
                                     : (vinn + ((size_t)gj * 3 + (mat - 2)) * DD);
      const bf16_t* pi = (i < ROWCAP) ? &rows[i * DD] : gsi;
      const bf16_t* pj = (j < ROWCAP) ? &rows[j * DD] : gsj;

      float dot = 0.0f;
      #pragma unroll 8
      for (int c = 0; c < DD; ++c) dot += (float)pi[c] * (float)pj[c];

      float tgt;
      if (mat == 0)      tgt = 0.4f;                                    // BRICS
      else if (mat == 1) tgt = (atypes[gi] == atypes[gj]) ? 0.3f : 0.1f;
      else               tgt = 0.2f;                                    // BRICS-0.2
      float d = dot - tgt;
      lsum += d * d;
    }
    float tot = block_reduce256(lsum, red);
    if (t == 0) res[mat] = tot;
    __syncthreads();
  }

  if (t == 0) {
    float c = (float)cnt;
    float pairs = c * (c - 1.0f) * 0.5f;
    float sp = fmaxf(pairs, 1.0f);
    sh_sim[f]   = res[0] / sp;
    un_sim[f]   = res[1] / sp;
    dir_loss[f] = (res[2] + res[3] + res[4]) / (3.0f * sp);
  }
}

// ---------------------------------------------------------------------------
// Final reduction across fragments
// ---------------------------------------------------------------------------
__global__ __launch_bounds__(256) void k_finalize(
    const int* __restrict__ counts, const float* __restrict__ sh_var,
    const float* __restrict__ v_var, const float* __restrict__ sh_sim,
    const float* __restrict__ un_sim, const float* __restrict__ dirl,
    const float* __restrict__ ss2, float* __restrict__ out) {
  __shared__ float redA[256];
  __shared__ float redB[256];
  const int f = threadIdx.x;
  const float c = (float)counts[f];
  const int valid = (c >= 2.0f);
  const float l2 = 0.01f * (sqrtf(ss2[0]) + sqrtf(ss2[1]));
  const float fl = 0.4f * un_sim[f] + 0.6f * (sh_var[f] + sh_sim[f]) +
                   0.2f * (v_var[f] + dirl[f]) + l2;
  redA[f] = valid ? fl : 0.0f;
  redB[f] = valid ? 1.0f : 0.0f;
  __syncthreads();
  #pragma unroll
  for (int s = 128; s > 0; s >>= 1) {
    if (f < s) { redA[f] += redA[f + s]; redB[f] += redB[f + s]; }
    __syncthreads();
  }
  if (f == 0) {
    float nv = redB[0];
    out[0] = (nv > 0.0f) ? 0.1f * redA[0] / fmaxf(nv, 1.0f) : 0.0f;
  }
}

// ---------------------------------------------------------------------------
extern "C" void kernel_launch(void* const* d_in, const int* in_sizes, int n_in,
                              void* d_out, int out_size, void* d_ws, size_t ws_size,
                              hipStream_t stream) {
  const float* scalar_short = (const float*)d_in[0];
  const float* scalar_long  = (const float*)d_in[1];
  const float* vector_short = (const float*)d_in[2];
  const float* vector_long  = (const float*)d_in[3];
  const float* W1  = (const float*)d_in[4];
  const float* b1  = (const float*)d_in[5];
  const float* W2  = (const float*)d_in[6];
  const float* b2  = (const float*)d_in[7];
  const float* Wd1 = (const float*)d_in[8];
  const float* bd1 = (const float*)d_in[9];
  const float* Wd2 = (const float*)d_in[10];
  const float* bd2 = (const float*)d_in[11];
  const float* Wv1 = (const float*)d_in[12];
  const float* Wv2 = (const float*)d_in[13];
  const int* frag  = (const int*)d_in[14];
  const int* atyp  = (const int*)d_in[15];
  float* out = (float*)d_out;

  char* ws = (char*)d_ws;
  size_t cur = 0;
  auto alloc = [&](size_t bytes) -> char* {
    char* p = ws + cur;
    cur += (bytes + 255) & ~(size_t)255;
    return p;
  };

  // pre-transposed bf16 weights (N x K)
  bf16_t* w1t  = (bf16_t*)alloc((size_t)DD * DD * 2);
  bf16_t* w2t  = (bf16_t*)alloc((size_t)DD * DD * 2);
  bf16_t* wd1t = (bf16_t*)alloc((size_t)DD * TDIM * 2);
  bf16_t* wd2t = (bf16_t*)alloc((size_t)TDIM * TDIM * 2);
  bf16_t* wv1t = (bf16_t*)alloc((size_t)DD * DD * 2);
  bf16_t* wv2t = (bf16_t*)alloc((size_t)DD * DD * 2);
  bf16_t* sc   = (bf16_t*)alloc((size_t)NN * DD * 2);
  bf16_t* vc   = (bf16_t*)alloc((size_t)NVEC * DD * 2);
  bf16_t* h1   = (bf16_t*)alloc((size_t)NN * DD * 2);
  bf16_t* sinv = (bf16_t*)alloc((size_t)NN * DD * 2);
  bf16_t* h2   = (bf16_t*)alloc((size_t)NN * TDIM * 2);
  bf16_t* hv   = (bf16_t*)alloc((size_t)NVEC * DD * 2);
  float*  dec  = (float*)alloc((size_t)NN * TDIM * 4);
  float*  vinv = (float*)alloc((size_t)NVEC * DD * 4);
  bf16_t* shn  = (bf16_t*)alloc((size_t)NN * DD * 2);
  bf16_t* unn  = (bf16_t*)alloc((size_t)NN * DD * 2);
  bf16_t* vinn = (bf16_t*)alloc((size_t)NVEC * DD * 2);
  int* counts  = (int*)alloc(NFRG * 4);
  int* cursor  = (int*)alloc(NFRG * 4);
  float* ss2   = (float*)alloc(2 * 4);
  int* offs    = (int*)alloc((NFRG + 1) * 4);
  int* members = (int*)alloc((size_t)NN * 4);
  float* shv   = (float*)alloc(NFRG * 4);
  float* vvr   = (float*)alloc(NFRG * 4);
  float* shs   = (float*)alloc(NFRG * 4);
  float* uns   = (float*)alloc(NFRG * 4);
  float* dirl  = (float*)alloc(NFRG * 4);
  (void)ws_size; (void)n_in; (void)in_sizes; (void)out_size;

  hipMemsetAsync(counts, 0, NFRG * 4, stream);
  hipMemsetAsync(cursor, 0, NFRG * 4, stream);
  hipMemsetAsync(ss2, 0, 2 * 4, stream);

  const int thr = 256;
  k_lerp_bf16<<<(NN * DD + thr - 1) / thr, thr, 0, stream>>>(
      scalar_short, scalar_long, sc, NN * DD, 0.3f);
  k_lerp_bf16<<<(NVEC * DD + thr - 1) / thr, thr, 0, stream>>>(
      vector_short, vector_long, vc, NVEC * DD, 0.3f);
  k_cvt_transpose_bf16<<<(DD * DD + thr - 1) / thr, thr, 0, stream>>>(W1, w1t, DD, DD);
  k_cvt_transpose_bf16<<<(DD * DD + thr - 1) / thr, thr, 0, stream>>>(W2, w2t, DD, DD);
  k_cvt_transpose_bf16<<<(DD * TDIM + thr - 1) / thr, thr, 0, stream>>>(Wd1, wd1t, DD, TDIM);
  k_cvt_transpose_bf16<<<(TDIM * TDIM + thr - 1) / thr, thr, 0, stream>>>(Wd2, wd2t, TDIM, TDIM);
  k_cvt_transpose_bf16<<<(DD * DD + thr - 1) / thr, thr, 0, stream>>>(Wv1, wv1t, DD, DD);
  k_cvt_transpose_bf16<<<(DD * DD + thr - 1) / thr, thr, 0, stream>>>(Wv2, wv2t, DD, DD);

  // scalar MLP chain
  gemm_bf16_wmma<<<dim3(DD / 128, NN / 64), thr, 0, stream>>>(
      sc, w1t, b1, h1, nullptr, NN, DD, DD, 1);
  gemm_bf16_wmma<<<dim3(DD / 128, NN / 64), thr, 0, stream>>>(
      h1, w2t, b2, sinv, nullptr, NN, DD, DD, 0);
  gemm_bf16_wmma<<<dim3(TDIM / 128, NN / 64), thr, 0, stream>>>(
      sinv, wd1t, bd1, h2, nullptr, NN, TDIM, DD, 1);
  gemm_bf16_wmma<<<dim3(TDIM / 128, NN / 64), thr, 0, stream>>>(
      h2, wd2t, bd2, nullptr, dec, NN, TDIM, TDIM, 0);
  // vector MLP chain
  gemm_bf16_wmma<<<dim3(DD / 128, NVEC / 64), thr, 0, stream>>>(
      vc, wv1t, nullptr, hv, nullptr, NVEC, DD, DD, 1);
  gemm_bf16_wmma<<<dim3(DD / 128, NVEC / 64), thr, 0, stream>>>(
      hv, wv2t, nullptr, nullptr, vinv, NVEC, DD, DD, 0);

  normalize_rows256<<<NN / 8, thr, 0, stream>>>(dec, TDIM, shn, DD, NN, &ss2[0]);
  normalize_rows256<<<NN / 8, thr, 0, stream>>>(dec + DD, TDIM, unn, DD, NN, &ss2[1]);
  normalize_rows256<<<NVEC / 8, thr, 0, stream>>>(vinv, DD, vinn, DD, NVEC, nullptr);

  k_count<<<NN / thr, thr, 0, stream>>>(frag, counts);
  k_scan<<<1, thr, 0, stream>>>(counts, offs, cursor);
  k_scatter<<<NN / thr, thr, 0, stream>>>(frag, cursor, members);

  frag_stats<<<NFRG, thr, 0, stream>>>(dec, vinv, members, offs, counts, shv, vvr);
  frag_pairs<<<NFRG, thr, 0, stream>>>(shn, unn, vinn, members, offs, counts,
                                       atyp, shs, uns, dirl);
  k_finalize<<<1, thr, 0, stream>>>(counts, shv, vvr, shs, uns, dirl, ss2, out);
}
